// Attention_86938728006171
// MI455X (gfx1250) — matrix operand order repaired and checked by reading, not compiled, not run on hardware
//
#include <hip/hip_runtime.h>

// ---- types matching gfx1250 WMMA builtin signatures ----
typedef __attribute__((ext_vector_type(16))) __bf16 v16bf;
typedef __attribute__((ext_vector_type(8)))  float  v8f;
typedef __attribute__((ext_vector_type(2)))  float  f32x2;
typedef __attribute__((ext_vector_type(2)))  __bf16 bf16x2;

#define B_      4
#define N_      2048
#define DM_     512
#define H_      8
#define DIMS_   64
#define SCALE_  0.044194173824159216f   // 1/sqrt(512)

static __device__ __forceinline__ unsigned short f2bf(float f) {
    __bf16 h = (__bf16)f;                    // backend picks v_cvt_pk_bf16_f32
    unsigned short u; __builtin_memcpy(&u, &h, 2); return u;
}

static __device__ __forceinline__ v16bf ld16(const unsigned short* p) {
    v16bf r; __builtin_memcpy(&r, p, 32); return r;     // 2x b128 loads
}
static __device__ __forceinline__ v16bf ld8x2(const unsigned short* p0, const unsigned short* p1) {
    v16bf r;
    __builtin_memcpy(&r, p0, 16);
    __builtin_memcpy(((char*)&r) + 16, p1, 16);
    return r;
}
static __device__ __forceinline__ v8f wmma_bf(v16bf a, v16bf b, v8f c) {
    return __builtin_amdgcn_wmma_f32_16x16x32_bf16(false, a, false, b, (short)0, c, false, false);
}

// ------------------------------------------------------------------
// 1) bf16 conversions: inputs -> Xb (paired, v_cvt_pk_bf16_f32),
//    attn_kernel -> Wqkvt[((c*8+h)*64+o)*512+i], out_kernel -> W2t[o*512+h*64+i]
// ------------------------------------------------------------------
#define NXPAIR_ (B_ * N_ * DM_ / 2)          // 2,097,152 float2 pairs
#define NW1_    (3 * 8 * 64 * 512)           // 786,432
#define NW2_    (64 * 512)                   // 32,768

__global__ __launch_bounds__(256) void prep_bf16(const float* __restrict__ in,
                                                 const float* __restrict__ ak,
                                                 const float* __restrict__ ok,
                                                 unsigned short* __restrict__ Xb,
                                                 unsigned short* __restrict__ Wqkvt,
                                                 unsigned short* __restrict__ W2t) {
    int idx = blockIdx.x * 256 + threadIdx.x;
    if (idx < NXPAIR_) {
        f32x2 v; __builtin_memcpy(&v, in + (size_t)idx * 2, 8);
        bf16x2 b = __builtin_convertvector(v, bf16x2);
        __builtin_memcpy(Xb + (size_t)idx * 2, &b, 4);
    } else if (idx < NXPAIR_ + NW1_) {
        int j = idx - NXPAIR_;
        int i = j & 511; int rest = j >> 9;
        int o = rest & 63; int ch = rest >> 6;          // ch = c*8+h
        int h = ch & 7;    int c = ch >> 3;
        Wqkvt[j] = f2bf(ak[(((i * 8 + h) * 64) + o) * 3 + c]);
    } else {
        int j = idx - NXPAIR_ - NW1_;
        if (j < NW2_) {
            int i = j & 63; int h = (j >> 6) & 7; int o = j >> 9;
            W2t[j] = f2bf(ok[((i * 8 + h) * 64) + o]);
        }
    }
}

// ------------------------------------------------------------------
// 2) fused QKV projection: 16x64 strip per wave, double-buffered K-loop
//    Q,K -> [BH, N, 64] bf16 ; V -> [BH, 64, N] bf16 (transposed)
// ------------------------------------------------------------------
__global__ __launch_bounds__(256) void qkv_gemm(const unsigned short* __restrict__ Xb,
                                                const unsigned short* __restrict__ Wqkvt,
                                                unsigned short* __restrict__ Qb,
                                                unsigned short* __restrict__ Kb,
                                                unsigned short* __restrict__ Vt) {
    int id = blockIdx.x * 8 + (threadIdx.x >> 5);        // 0..12287
    int lane = threadIdx.x & 31, ln = lane & 15, hi = lane >> 4;
    int ch = id % 24; int mt = id / 24;                  // ch = c*8+h, mt: 512 row tiles
    int c = ch >> 3, h = ch & 7;

    const unsigned short* arow = Xb + (size_t)(mt * 16 + ln) * DM_ + 16 * hi;
    const unsigned short* bbase = Wqkvt + (size_t)ch * 64 * DM_ + (size_t)ln * DM_ + 16 * hi;

    v8f acc[4];
#pragma unroll
    for (int t = 0; t < 4; ++t)
#pragma unroll
        for (int r = 0; r < 8; ++r) acc[t][r] = 0.f;

    // ---- software-pipelined K loop: load k+1 while computing k ----
    v16bf a_cur = ld8x2(arow, arow + 8);
    v16bf b_cur[4];
#pragma unroll
    for (int t = 0; t < 4; ++t) b_cur[t] = ld16(bbase + (size_t)t * 16 * DM_);

#pragma unroll
    for (int kk = 0; kk < 16; ++kk) {
        v16bf a_nxt = a_cur;
        v16bf b_nxt[4];
        if (kk < 15) {
            const unsigned short* p0 = arow + (kk + 1) * 32;
            a_nxt = ld8x2(p0, p0 + 8);
#pragma unroll
            for (int t = 0; t < 4; ++t)
                b_nxt[t] = ld16(bbase + (size_t)t * 16 * DM_ + (kk + 1) * 32);
        } else {
#pragma unroll
            for (int t = 0; t < 4; ++t) b_nxt[t] = b_cur[t];
        }
#pragma unroll
        for (int t = 0; t < 4; ++t)
            acc[t] = wmma_bf(a_cur, b_cur[t], acc[t]);
        a_cur = a_nxt;
#pragma unroll
        for (int t = 0; t < 4; ++t) b_cur[t] = b_nxt[t];
    }

#pragma unroll
    for (int t = 0; t < 4; ++t)
#pragma unroll
        for (int r = 0; r < 8; ++r) {                    // C/D: M = r + 8*hi, N = ln
            int grow = mt * 16 + r + 8 * hi;
            int bidx = grow >> 11, nrow = grow & 2047;
            int o = t * 16 + ln;
            unsigned short val = f2bf(acc[t][r]);
            size_t bh = (size_t)bidx * 8 + h;
            if (c == 0)      Kb[(bh * N_ + nrow) * DIMS_ + o] = val;
            else if (c == 2) Qb[(bh * N_ + nrow) * DIMS_ + o] = val;
            else             Vt[(bh * DIMS_ + o) * N_ + nrow] = val;
        }
}

// ------------------------------------------------------------------
// 3) causal flash attention: one 16-query tile per wave, 32-key chunks,
//    online softmax; V-fragment loads hoisted over the softmax phase
// ------------------------------------------------------------------
__global__ __launch_bounds__(256) void attn(const unsigned short* __restrict__ Qb,
                                            const unsigned short* __restrict__ Kb,
                                            const unsigned short* __restrict__ Vt,
                                            unsigned short* __restrict__ OHb) {
    __shared__ unsigned short plds[8][16][40];           // per-wave 16x32 + pad
    int wv = threadIdx.x >> 5;
    int lane = threadIdx.x & 31, ln = lane & 15, hi = lane >> 4;
    int bh = blockIdx.x >> 4, qblk = blockIdx.x & 15;
    int b = bh >> 3, h = bh & 7;
    int qbase = qblk * 128 + wv * 16;

    const unsigned short* qrow = Qb + ((size_t)bh * N_ + qbase + ln) * DIMS_;
    v16bf qf0 = ld8x2(qrow + 16 * hi,      qrow + 8 + 16 * hi);   // d = 0..31
    v16bf qf1 = ld8x2(qrow + 32 + 16 * hi, qrow + 40 + 16 * hi);   // d = 32..63

    float m_[8], l_[8]; v8f oacc[4];
#pragma unroll
    for (int r = 0; r < 8; ++r) { m_[r] = -1e30f; l_[r] = 0.f; }
#pragma unroll
    for (int t = 0; t < 4; ++t)
#pragma unroll
        for (int r = 0; r < 8; ++r) oacc[t][r] = 0.f;

    int nch = (qbase + 47) >> 5;                         // keys 0..qbase+15 covered
    for (int cblk = 0; cblk < nch; ++cblk) {
        int kbase = cblk * 32;
        const unsigned short* kptr = Kb + ((size_t)bh * N_ + kbase) * DIMS_;

        v8f s0 = {0.f,0.f,0.f,0.f,0.f,0.f,0.f,0.f};
        v8f s1 = {0.f,0.f,0.f,0.f,0.f,0.f,0.f,0.f};
        // S0: keys kbase..+15 , S1: keys kbase+16..+31 ; contraction d in 2 steps
        s0 = wmma_bf(qf0, ld16(kptr + (size_t)ln * DIMS_ + 16 * hi), s0);
        s1 = wmma_bf(qf0, ld16(kptr + (size_t)(16 + ln) * DIMS_ + 16 * hi), s1);
        s0 = wmma_bf(qf1, ld16(kptr + (size_t)ln * DIMS_ + 32 + 16 * hi), s0);
        s1 = wmma_bf(qf1, ld16(kptr + (size_t)(16 + ln) * DIMS_ + 32 + 16 * hi), s1);

        // V fragments: independent of softmax -> issue now, consume after it
        v16bf vb[4];
#pragma unroll
        for (int t = 0; t < 4; ++t)
            vb[t] = ld16(Vt + ((size_t)bh * DIMS_ + t * 16 + ln) * N_ + kbase + 16 * hi);

        if (cblk + 1 < nch)                              // prefetch next K chunk
            __builtin_prefetch(kptr + (size_t)(32 + ln) * DIMS_, 0, 1);

#pragma unroll
        for (int r = 0; r < 8; ++r) {
            int row = qbase + r + 8 * hi;
            float f0 = s0[r] * SCALE_; if (kbase + ln > row)      f0 = -1e30f;
            float f1 = s1[r] * SCALE_; if (kbase + 16 + ln > row) f1 = -1e30f;
            float cm = fmaxf(f0, f1);
            cm = fmaxf(cm, __shfl_xor(cm, 1));
            cm = fmaxf(cm, __shfl_xor(cm, 2));
            cm = fmaxf(cm, __shfl_xor(cm, 4));
            cm = fmaxf(cm, __shfl_xor(cm, 8));           // row max within 16-lane half
            float mnew = fmaxf(m_[r], cm);
            float alpha = __expf(m_[r] - mnew);
            m_[r] = mnew;
            float p0 = __expf(f0 - mnew);
            float p1 = __expf(f1 - mnew);
            float rs = p0 + p1;
            rs += __shfl_xor(rs, 1);
            rs += __shfl_xor(rs, 2);
            rs += __shfl_xor(rs, 4);
            rs += __shfl_xor(rs, 8);
            l_[r] = l_[r] * alpha + rs;
#pragma unroll
            for (int t = 0; t < 4; ++t) oacc[t][r] *= alpha;
            plds[wv][r + 8 * hi][ln]      = f2bf(p0);    // C-layout -> LDS tile
            plds[wv][r + 8 * hi][16 + ln] = f2bf(p1);
        }
        asm volatile("s_wait_dscnt 0x0" ::: "memory");
        // reload P as A-fragment (M=ln, K = 16*hi+e)
        v16bf pA = ld8x2(&plds[wv][ln][16 * hi], &plds[wv][ln][8 + 16 * hi]);
#pragma unroll
        for (int t = 0; t < 4; ++t)
            oacc[t] = wmma_bf(pA, vb[t], oacc[t]);
    }

#pragma unroll
    for (int r = 0; r < 8; ++r) {
        float inv = 1.0f / l_[r];
        int row = qbase + r + 8 * hi;
        size_t base = (((size_t)b * N_ + row) * H_ + h) * DIMS_;
#pragma unroll
        for (int t = 0; t < 4; ++t)
            OHb[base + t * 16 + ln] = f2bf(oacc[t][r] * inv);
    }
}

// ------------------------------------------------------------------
// 4) output projection: [8192 x 512] x [512 x 64] -> f32 out
//    16x64 strip per wave, double-buffered K-loop
// ------------------------------------------------------------------
__global__ __launch_bounds__(256) void out_gemm(const unsigned short* __restrict__ OHb,
                                                const unsigned short* __restrict__ W2t,
                                                float* __restrict__ out) {
    int id = blockIdx.x * 8 + (threadIdx.x >> 5);        // 0..511 (row tiles)
    int lane = threadIdx.x & 31, ln = lane & 15, hi = lane >> 4;
    int mt = id;

    const unsigned short* arow = OHb + (size_t)(mt * 16 + ln) * DM_ + 16 * hi;
    const unsigned short* bbase = W2t + (size_t)ln * DM_ + 16 * hi;

    v8f acc[4];
#pragma unroll
    for (int t = 0; t < 4; ++t)
#pragma unroll
        for (int r = 0; r < 8; ++r) acc[t][r] = 0.f;

    v16bf a_cur = ld8x2(arow, arow + 8);
    v16bf b_cur[4];
#pragma unroll
    for (int t = 0; t < 4; ++t) b_cur[t] = ld16(bbase + (size_t)t * 16 * DM_);

#pragma unroll
    for (int kk = 0; kk < 16; ++kk) {
        v16bf a_nxt = a_cur;
        v16bf b_nxt[4];
        if (kk < 15) {
            const unsigned short* p0 = arow + (kk + 1) * 32;
            a_nxt = ld8x2(p0, p0 + 8);
#pragma unroll
            for (int t = 0; t < 4; ++t)
                b_nxt[t] = ld16(bbase + (size_t)t * 16 * DM_ + (kk + 1) * 32);
        } else {
#pragma unroll
            for (int t = 0; t < 4; ++t) b_nxt[t] = b_cur[t];
        }
#pragma unroll
        for (int t = 0; t < 4; ++t)
            acc[t] = wmma_bf(a_cur, b_cur[t], acc[t]);
        a_cur = a_nxt;
#pragma unroll
        for (int t = 0; t < 4; ++t) b_cur[t] = b_nxt[t];
    }

#pragma unroll
    for (int t = 0; t < 4; ++t)
#pragma unroll
        for (int r = 0; r < 8; ++r)
            out[(size_t)(mt * 16 + r + 8 * hi) * DIMS_ + t * 16 + ln] = acc[t][r];
}

// ------------------------------------------------------------------
extern "C" void kernel_launch(void* const* d_in, const int* in_sizes, int n_in,
                              void* d_out, int out_size, void* d_ws, size_t ws_size,
                              hipStream_t stream) {
    (void)in_sizes; (void)n_in; (void)out_size; (void)ws_size;
    const float* in = (const float*)d_in[0];
    const float* ak = (const float*)d_in[1];
    const float* ok = (const float*)d_in[2];
    float* out = (float*)d_out;

    char* ws = (char*)d_ws;
    const size_t SZ = (size_t)B_ * H_ * N_ * DIMS_ * sizeof(unsigned short); // 8 MiB
    unsigned short* Qb    = (unsigned short*)(ws);
    unsigned short* Kb    = (unsigned short*)(ws + SZ);
    unsigned short* Vt    = (unsigned short*)(ws + 2 * SZ);
    unsigned short* OHb   = (unsigned short*)(ws + 3 * SZ);
    unsigned short* Wqkvt = (unsigned short*)(ws + 4 * SZ);
    unsigned short* W2t   = (unsigned short*)(ws + 4 * SZ + (size_t)NW1_ * 2);
    unsigned short* Xb    = (unsigned short*)(ws + 4 * SZ + (size_t)NW1_ * 2 + (size_t)NW2_ * 2);

    int prep_threads = NXPAIR_ + NW1_ + NW2_;
    prep_bf16<<<(prep_threads + 255) / 256, 256, 0, stream>>>(in, ak, ok, Xb, Wqkvt, W2t);
    qkv_gemm<<<1536, 256, 0, stream>>>(Xb, Wqkvt, Qb, Kb, Vt);
    attn<<<512, 256, 0, stream>>>(Qb, Kb, Vt, OHb);
    out_gemm<<<64, 256, 0, stream>>>(OHb, W2t, out);
}
